// Update_14370960572696
// MI455X (gfx1250) — matrix operand order repaired and checked by reading, not compiled, not run on hardware
//
#include <hip/hip_runtime.h>

// ---------------- problem constants (from reference) ----------------
#define DIMV   384
#define MV     96
#define GV     768
#define EV     (MV * GV)      // 73728
#define TDIMV  512
#define NHV    8
#define HDV    64
#define FFV    2048
#define CORRV  882
#define LNEPS  1e-3f

// flags for GEMM epilogue
#define F_RELU 1
#define F_SIG  2
#define F_ACC  4

typedef __bf16 v16bf __attribute__((ext_vector_type(16)));
typedef float  v8f   __attribute__((ext_vector_type(8)));

union FragBF {
  v16bf v;
  unsigned u[8];
  unsigned short h[16];
};

__device__ inline unsigned short f2bfu(float f) {
  unsigned u = __float_as_uint(f);
  return (unsigned short)((u + 0x7FFFu + ((u >> 16) & 1u)) >> 16);  // RNE
}

// ---------------- fp32 -> padded bf16 conversion (weights & raw activations) ----------------
__global__ void k_cvtw(unsigned short* __restrict__ dst, const float* __restrict__ src,
                       int N, int K, int Kp) {
  long long i = (long long)blockIdx.x * blockDim.x + threadIdx.x;
  if (i >= (long long)N * Kp) return;
  int n = (int)(i / Kp), k = (int)(i % Kp);
  dst[i] = (k < K) ? f2bfu(src[(size_t)n * K + k]) : (unsigned short)0;
}

// ---------------- WMMA GEMM:  C[M,N] = ep(Abf16[M,Kp] * Wbf16[N,Kp]^T + bias) ----------------
// Block tile 128x128, BK=32, double-buffered LDS. 8 waves as 2x4; each wave owns 64x32
// (4x2 accumulators) -> 8 x v_wmma_f32_16x16x32_bf16 per K-step between barriers.
// A loads are UNCONDITIONAL: callers guarantee A has capacity for gridDim.y*128 rows.
__global__ __launch_bounds__(256) void k_gemm(
    const unsigned short* __restrict__ A,   // bf16 [>=ceil(Mr/128)*128 x Kp]
    const unsigned short* __restrict__ W,   // bf16 [N x Kp], N multiple of 128
    const float* __restrict__ bias,
    float* __restrict__ C,                  // fp32 out (nullable)
    unsigned short* __restrict__ Cb,        // bf16 out (nullable), stride N
    int Mr, int N, int Kp, int flags)
{
  __shared__ unsigned short As[2][128][32];
  __shared__ unsigned short Bs[2][128][32];
  const int tid  = threadIdx.x;
  const int wave = tid >> 5, lane = tid & 31;
  const int wm   = wave >> 2, wn = wave & 3;     // 2 x 4 wave grid (64 rows x 32 cols each)
  const int bm   = blockIdx.y * 128, bn = blockIdx.x * 128;
  const int l15  = lane & 15, kh = lane >> 4;

  // tile staging: 128 rows x 64B = 512 16B-chunks; thread handles chunk tid and tid+256
  const int r0 = tid >> 2, sg = (tid & 3) * 8;
  const unsigned short* Aptr = A + (size_t)(bm + r0) * Kp + sg;
  const unsigned short* Bptr = W + (size_t)(bn + r0) * Kp + sg;
  const size_t rowskip = (size_t)64 * Kp;

  int4 pa0, pa1, pb0, pb1;
  auto prefetch = [&](int k0) {
    pa0 = *reinterpret_cast<const int4*>(Aptr + k0);
    pa1 = *reinterpret_cast<const int4*>(Aptr + rowskip + k0);
    pb0 = *reinterpret_cast<const int4*>(Bptr + k0);
    pb1 = *reinterpret_cast<const int4*>(Bptr + rowskip + k0);
  };
  auto stage = [&](int buf) {
    *reinterpret_cast<int4*>(&As[buf][r0][sg])      = pa0;
    *reinterpret_cast<int4*>(&As[buf][r0 + 64][sg]) = pa1;
    *reinterpret_cast<int4*>(&Bs[buf][r0][sg])      = pb0;
    *reinterpret_cast<int4*>(&Bs[buf][r0 + 64][sg]) = pb1;
  };

  v8f acc[4][2] = {};
  const int nk = Kp >> 5;
  prefetch(0);
  stage(0);
  __syncthreads();
  for (int i = 0; i < nk; ++i) {
    const int buf = i & 1;
    if (i + 1 < nk) prefetch((i + 1) << 5);
#pragma unroll
    for (int tm = 0; tm < 4; ++tm) {
      FragBF fa;
#pragma unroll
      for (int j = 0; j < 8; ++j) {
        int ka = ((j < 4) ? 0 : 16) + kh * 8 + (j & 3) * 2;
        fa.u[j] = *reinterpret_cast<const unsigned*>(&As[buf][wm * 64 + tm * 16 + l15][ka]);
      }
#pragma unroll
      for (int tn = 0; tn < 2; ++tn) {
        FragBF fb;
#pragma unroll
        for (int j = 0; j < 8; ++j) {
          int kb = ((j < 4) ? 0 : 16) + kh * 8 + (j & 3) * 2;
          fb.u[j] = *reinterpret_cast<const unsigned*>(&Bs[buf][wn * 32 + tn * 16 + l15][kb]);
        }
        acc[tm][tn] = __builtin_amdgcn_wmma_f32_16x16x32_bf16(false, fa.v, false, fb.v,
                                                              (short)0, acc[tm][tn],
                                                              false, false);
      }
    }
    if (i + 1 < nk) { stage(buf ^ 1); __syncthreads(); }
  }

#pragma unroll
  for (int tn = 0; tn < 2; ++tn) {
    const int col = bn + wn * 32 + tn * 16 + l15;
    const float bv = bias ? bias[col] : 0.0f;
#pragma unroll
    for (int tm = 0; tm < 4; ++tm) {
#pragma unroll
      for (int j = 0; j < 8; ++j) {
        int row = bm + wm * 64 + tm * 16 + kh * 8 + j;
        if (row < Mr) {
          float r = acc[tm][tn][j] + bv;
          if (flags & F_RELU) r = fmaxf(r, 0.0f);
          if (flags & F_SIG)  r = 1.0f / (1.0f + __expf(-r));
          size_t o = (size_t)row * N + col;
          if (flags & F_ACC) { r += C[o]; C[o] = r; }
          else if (C)        { C[o] = r; }
          if (Cb) Cb[o] = f2bfu(r);
        }
      }
    }
  }
}

// ---------------- fused attention per (group, head): softmax(QK^T/8) V ----------------
// qkv already bf16 [E,1536]; output written bf16-only for out_proj.
// LDS: [0,24K) Q,K bf16 -> later S bf16 ; [24K,60K) S fp32 -> later V bf16.
__global__ __launch_bounds__(256) void k_attn(const unsigned short* __restrict__ qkvb,
                                              unsigned short* __restrict__ ob)
{
  __shared__ __align__(16) char smem[24 * 1024 + 36 * 1024];
  unsigned short* Qs = (unsigned short*)smem;               // 96*64
  unsigned short* Ks = Qs + 96 * 64;                        // 96*64
  unsigned short* Sb = (unsigned short*)smem;               // 96*96 (after Q/K dead)
  float*          Sf = (float*)(smem + 24 * 1024);          // 96*96
  unsigned short* Vs = (unsigned short*)(smem + 24 * 1024); // 96*64 (after Sf dead)

  const int gh = blockIdx.x;
  const int g = gh / NHV, h = gh % NHV;
  const int tid = threadIdx.x, wave = tid >> 5, lane = tid & 31;
  const int l15 = lane & 15, kh = lane >> 4;
  const size_t base = (size_t)g * MV * (3 * TDIMV);

  for (int i = tid; i < MV * HDV; i += 256) {
    int m = i >> 6, d = i & 63;
    size_t rb = base + (size_t)m * (3 * TDIMV) + h * HDV + d;
    Qs[i] = qkvb[rb];
    Ks[i] = qkvb[rb + TDIMV];
  }
  __syncthreads();

  for (int t = wave; t < 36; t += 8) {     // 6x6 score tiles, K=64 in two chunks
    int tm = t / 6, tn = t % 6;
    v8f acc = {};
    for (int kc = 0; kc < HDV; kc += 32) {
      FragBF fa, fb;
#pragma unroll
      for (int i = 0; i < 8; ++i) {
        int ka = kc + ((i < 4) ? 0 : 16) + kh * 8 + (i & 3) * 2;
        fa.u[i] = *reinterpret_cast<const unsigned*>(&Qs[(tm * 16 + l15) * HDV + ka]);
        int kb = kc + ((i < 4) ? 0 : 16) + kh * 8 + (i & 3) * 2;
        fb.u[i] = *reinterpret_cast<const unsigned*>(&Ks[(tn * 16 + l15) * HDV + kb]);
      }
      acc = __builtin_amdgcn_wmma_f32_16x16x32_bf16(false, fa.v, false, fb.v,
                                                    (short)0, acc, false, false);
    }
#pragma unroll
    for (int i = 0; i < 8; ++i) {
      int row = tm * 16 + kh * 8 + i, col = tn * 16 + l15;
      Sf[row * MV + col] = acc[i] * 0.125f;   // 1/sqrt(64)
    }
  }
  __syncthreads();

  if (tid < MV) {   // softmax, one thread per row; write probs as bf16 into Sb
    float mx = -1e30f;
    for (int c = 0; c < MV; ++c) mx = fmaxf(mx, Sf[tid * MV + c]);
    float s = 0.0f;
    for (int c = 0; c < MV; ++c) s += __expf(Sf[tid * MV + c] - mx);
    float inv = 1.0f / s;
    for (int c = 0; c < MV; ++c)
      Sb[tid * MV + c] = f2bfu(__expf(Sf[tid * MV + c] - mx) * inv);
  }
  __syncthreads();

  for (int i = tid; i < MV * HDV; i += 256) {   // V overlays Sf region
    int k = i >> 6, d = i & 63;
    Vs[i] = qkvb[base + (size_t)k * (3 * TDIMV) + 2 * TDIMV + h * HDV + d];
  }
  __syncthreads();

  for (int t = wave; t < 24; t += 8) {   // O = Sb(96x96) x V(96x64), K=96 in 3 chunks
    int tm = t / 4, tn = t % 4;
    v8f acc = {};
    for (int kc = 0; kc < MV; kc += 32) {
      FragBF fa, fb;
#pragma unroll
      for (int i = 0; i < 8; ++i) {
        int ka = kc + ((i < 4) ? 0 : 16) + kh * 8 + (i & 3) * 2;
        fa.u[i] = *reinterpret_cast<const unsigned*>(&Sb[(tm * 16 + l15) * MV + ka]);
        int kb = kc + ((i < 4) ? 0 : 16) + kh * 8 + (i & 3) * 2;
        fb.h[2 * i]     = Vs[(size_t)kb * HDV + tn * 16 + l15];
        fb.h[2 * i + 1] = Vs[(size_t)(kb + 1) * HDV + tn * 16 + l15];
      }
      acc = __builtin_amdgcn_wmma_f32_16x16x32_bf16(false, fa.v, false, fb.v,
                                                    (short)0, acc, false, false);
    }
#pragma unroll
    for (int i = 0; i < 8; ++i) {
      int row = tm * 16 + kh * 8 + i, col = tn * 16 + l15;
      ob[((size_t)g * MV + row) * TDIMV + h * HDV + col] = f2bfu(acc[i]);
    }
  }
}

// ---------------- LayerNorm (optional x+add, optional ReLU, fp32 + bf16 outputs) ----------------
__global__ __launch_bounds__(256) void k_ln(
    float* __restrict__ out, unsigned short* __restrict__ outb,
    const float* __restrict__ in, const float* __restrict__ add,
    const float* __restrict__ gg, const float* __restrict__ bb, int rows, int dim, int relu)
{
  int row  = blockIdx.x * 8 + (threadIdx.x >> 5);
  int lane = threadIdx.x & 31;
  if (row >= rows) return;
  const float* x = in + (size_t)row * dim;
  const float* a = add ? add + (size_t)row * dim : nullptr;
  float vals[16];
  int n = dim >> 5;
  float s = 0.0f;
  for (int i = 0; i < n; ++i) {
    float v = x[lane + (i << 5)];
    if (a) v += a[lane + (i << 5)];
    vals[i] = v; s += v;
  }
  for (int off = 16; off; off >>= 1) s += __shfl_xor(s, off);
  float mu = s / dim;
  float vs = 0.0f;
  for (int i = 0; i < n; ++i) { float d = vals[i] - mu; vs += d * d; }
  for (int off = 16; off; off >>= 1) vs += __shfl_xor(vs, off);
  float inv = rsqrtf(vs / dim + LNEPS);
  for (int i = 0; i < n; ++i) {
    int c = lane + (i << 5);
    float r = (vals[i] - mu) * inv * gg[c] + bb[c];
    if (relu) r = fmaxf(r, 0.0f);
    if (out)  out[(size_t)row * dim + c] = r;
    if (outb) outb[(size_t)row * dim + c] = f2bfu(r);
  }
}

// ---------------- elementwise / gather / segment kernels ----------------
__global__ void k_add2(float* __restrict__ o, const float* __restrict__ a,
                       const float* __restrict__ b, long long n) {
  long long i = (long long)blockIdx.x * blockDim.x + threadIdx.x;
  if (i < n) o[i] = a[i] + b[i];
}

__global__ void k_gather(unsigned short* __restrict__ dst, const float* __restrict__ src,
                         const int* __restrict__ idx, long long rows, int dim) {
  long long i = (long long)blockIdx.x * blockDim.x + threadIdx.x;
  if (i >= rows * dim) return;
  long long e = i / dim; int c = (int)(i % dim);
  int s = idx[e];
  dst[i] = (s >= 0) ? f2bfu(src[(size_t)s * dim + c]) : (unsigned short)0;
}

__global__ void k_segmax(const float* __restrict__ gx, float* __restrict__ smax) {
  int t = blockIdx.x * blockDim.x + threadIdx.x;
  if (t >= MV * DIMV) return;
  int m = t / DIMV, c = t % DIMV;
  float mx = -1e30f;
  for (int g = 0; g < GV; ++g)
    mx = fmaxf(mx, gx[((size_t)g * MV + m) * DIMV + c]);
  smax[t] = mx;
}

__global__ void k_segsum(const float* __restrict__ gx, const float* __restrict__ fx,
                         const float* __restrict__ smax, unsigned short* __restrict__ yb) {
  int t = blockIdx.x * blockDim.x + threadIdx.x;
  if (t >= MV * DIMV) return;
  int m = t / DIMV, c = t % DIMV;
  float mx = smax[t], num = 0.0f, den = 0.0f;
  for (int g = 0; g < GV; ++g) {
    size_t o = ((size_t)g * MV + m) * DIMV + c;
    float e = __expf(gx[o] - mx);
    den += e; num += fx[o] * e;
  }
  yb[t] = f2bfu(num / den);
}

__global__ void k_bcast_add(float* __restrict__ net, unsigned short* __restrict__ netb,
                            const float* __restrict__ hy, long long n) {
  long long i = (long long)blockIdx.x * blockDim.x + threadIdx.x;
  if (i >= n) return;
  long long e = i / DIMV; int c = (int)(i % DIMV);
  float v = net[i] + hy[(e % MV) * DIMV + c];
  net[i] = v;
  netb[i] = f2bfu(v);
}

__global__ void k_gru_comb(float* __restrict__ net, const float* __restrict__ h,
                           const float* __restrict__ gate, const float* __restrict__ res,
                           long long n) {
  long long i = (long long)blockIdx.x * blockDim.x + threadIdx.x;
  if (i < n) net[i] = h[i] + gate[i] * res[i];
}

// ---------------- output heads: d (2), w, w_gn from relu(net) ----------------
__global__ __launch_bounds__(256) void k_heads(
    const float* __restrict__ net,
    const float* __restrict__ wd, const float* __restrict__ bd,
    const float* __restrict__ ww, const float* __restrict__ bw,
    const float* __restrict__ wg, const float* __restrict__ bg,
    float* __restrict__ out)
{
  int e = blockIdx.x * 8 + (threadIdx.x >> 5);
  int lane = threadIdx.x & 31;
  if (e >= EV) return;
  const float* x = net + (size_t)e * DIMV;
  float s0 = 0, s1 = 0, s2 = 0, s3 = 0;
  for (int i = lane; i < DIMV; i += 32) {
    float v = fmaxf(x[i], 0.0f);
    s0 += v * wd[i]; s1 += v * wd[DIMV + i]; s2 += v * ww[i]; s3 += v * wg[i];
  }
  for (int off = 16; off; off >>= 1) {
    s0 += __shfl_xor(s0, off); s1 += __shfl_xor(s1, off);
    s2 += __shfl_xor(s2, off); s3 += __shfl_xor(s3, off);
  }
  if (lane == 0) {
    out[(size_t)EV * DIMV + (size_t)e * 2 + 0] = s0 + bd[0];
    out[(size_t)EV * DIMV + (size_t)e * 2 + 1] = s1 + bd[1];
    out[(size_t)EV * 386 + e] = 1.0f / (1.0f + __expf(-(s2 + bw[0])));
    out[(size_t)EV * 387 + e] = 1.0f / (1.0f + __expf(-(s3 + bg[0])));
  }
}

// ---------------- host orchestration ----------------
struct WInfo { int idx; int N; int K; int Kp; };

extern "C" void kernel_launch(void* const* d_in, const int* in_sizes, int n_in,
                              void* d_out, int out_size, void* d_ws, size_t ws_size,
                              hipStream_t stream) {
  (void)in_sizes; (void)n_in; (void)out_size; (void)ws_size;
#define PP(i) ((const float*)d_in[(i)])
  const float* net_in   = PP(0);
  const float* inp      = PP(1);
  const float* corr     = PP(2);
  const float* residual = PP(3);
  const int*   ix = (const int*)d_in[7];
  const int*   jx = (const int*)d_in[8];
  float* netp = (float*)d_out;    // first E*384 floats of output == final net

  static const WInfo wt[27] = {
    {9,  DIMV, CORRV, 896},               // 0  corr.l1
    {11, DIMV, DIMV, DIMV},               // 1  corr.l2
    {15, DIMV, DIMV, DIMV},               // 2  corr.l3
    {17, DIMV, 2,    32},                 // 3  res.l1
    {21, DIMV, DIMV, DIMV},               // 4  res.l2
    {25, DIMV, DIMV, DIMV}, {27, DIMV, DIMV, DIMV},   // 5,6  c1
    {29, DIMV, DIMV, DIMV}, {31, DIMV, DIMV, DIMV},   // 7,8  c2
    {33, DIMV, DIMV, DIMV}, {35, DIMV, DIMV, DIMV}, {37, DIMV, DIMV, DIMV}, // 9-11 agg
    {39, TDIMV, DIMV, DIMV},              // 12 t_in
    {41, 3 * TDIMV, TDIMV, TDIMV},        // 13 in_proj
    {43, TDIMV, TDIMV, TDIMV},            // 14 out_proj
    {47, FFV, TDIMV, TDIMV},              // 15 lin1
    {49, TDIMV, FFV, FFV},                // 16 lin2
    {53, DIMV, TDIMV, TDIMV},             // 17 t_out
    {57, DIMV, DIMV, DIMV}, {59, DIMV, DIMV, DIMV}, {61, DIMV, DIMV, DIMV}, // gru0
    {65, DIMV, DIMV, DIMV}, {67, DIMV, DIMV, DIMV}, {69, DIMV, DIMV, DIMV}, // gru1
    {73, DIMV, DIMV, DIMV}, {75, DIMV, DIMV, DIMV}, {77, DIMV, DIMV, DIMV}, // gru2
  };
  size_t woff[27]; size_t wo = 0;
  for (int i = 0; i < 27; ++i) { woff[i] = wo; wo += (size_t)wt[i].N * wt[i].Kp; }

  // ---- workspace carve-up ----
  char* p = (char*)d_ws;
  auto carve = [&](size_t bytes) { char* r = p; p += (bytes + 255) & ~(size_t)255; return r; };
  unsigned short* WB     = (unsigned short*)carve(16ull << 20);                   // bf16 weights
  unsigned short* corr_b = (unsigned short*)carve((size_t)EV * 896 * 2);
  unsigned short* res_b  = (unsigned short*)carve((size_t)EV * 32 * 2);
  unsigned short* t0b    = (unsigned short*)carve((size_t)EV * 2048 * 2);         // FF hidden etc.
  unsigned short* t1b    = (unsigned short*)carve((size_t)EV * 1536 * 2);         // qkv etc.
  unsigned short* t2b    = (unsigned short*)carve((size_t)EV * 512 * 2);          // x / gru hidden
  unsigned short* ob     = (unsigned short*)carve((size_t)EV * 512 * 2);          // attn out
  unsigned short* netb   = (unsigned short*)carve((size_t)EV * DIMV * 2);         // net snapshot
  unsigned short* yb     = (unsigned short*)carve((size_t)128 * DIMV * 2);        // 128-row padded
  float* smax = (float*)carve((size_t)MV * DIMV * 4);
  float* hy   = (float*)carve((size_t)MV * DIMV * 4);
  float* t0   = (float*)carve((size_t)EV * DIMV * 4);
  float* t1   = (float*)carve((size_t)EV * 768 * 4);
  float* t2   = (float*)carve((size_t)EV * TDIMV * 4);
  float* t3   = (float*)carve((size_t)EV * DIMV * 4);

  auto nb = [](long long n) { return (unsigned)((n + 255) / 256); };

  for (int i = 0; i < 27; ++i) {
    long long n = (long long)wt[i].N * wt[i].Kp;
    k_cvtw<<<nb(n), 256, 0, stream>>>(WB + woff[i], PP(wt[i].idx),
                                      wt[i].N, wt[i].K, wt[i].Kp);
  }
  k_cvtw<<<nb((long long)EV * 896), 256, 0, stream>>>(corr_b, corr, EV, CORRV, 896);
  k_cvtw<<<nb((long long)EV * 32), 256, 0, stream>>>(res_b, residual, EV, 2, 32);

  auto gemm = [&](const unsigned short* A, int wi, float* C, unsigned short* Cb,
                  int Mr, int flags) {
    const WInfo& w = wt[wi];
    dim3 grid(w.N / 128, (Mr + 127) / 128);
    k_gemm<<<grid, 256, 0, stream>>>(A, WB + woff[wi], PP(w.idx + 1), C, Cb,
                                     Mr, w.N, w.Kp, flags);
  };
  auto ln = [&](float* o, unsigned short* obf, const float* in, const float* add,
                const float* g, const float* b, int rows, int dim, int relu) {
    k_ln<<<(rows + 7) / 8, 256, 0, stream>>>(o, obf, in, add, g, b, rows, dim, relu);
  };

  const long long ED = (long long)EV * DIMV;

  // ---- correlation encoder: net = net + inp + l3(relu(ln(l2(relu(l1(corr)))))) ----
  gemm(corr_b, 0, nullptr, t0b, EV, F_RELU);
  gemm(t0b, 1, t1, nullptr, EV, 0);
  ln(nullptr, t1b, t1, nullptr, PP(13), PP(14), EV, DIMV, 1);
  k_add2<<<nb(ED), 256, 0, stream>>>(netp, net_in, inp, ED);
  gemm(t1b, 2, netp, nullptr, EV, F_ACC);

  // ---- residual encoder: net += l2(relu(ln(l1(residual)))) ----
  gemm(res_b, 3, t0, nullptr, EV, 0);
  ln(nullptr, t0b, t0, nullptr, PP(19), PP(20), EV, DIMV, 1);
  gemm(t0b, 4, netp, nullptr, EV, F_ACC);

  // ---- net = LN(net) ----
  ln(netp, nullptr, netp, nullptr, PP(23), PP(24), EV, DIMV, 0);

  // ---- masked neighbor message passing ----
  k_gather<<<nb(ED), 256, 0, stream>>>(t0b, netp, ix, EV, DIMV);
  gemm(t0b, 5, nullptr, t1b, EV, F_RELU);
  gemm(t1b, 6, netp, nullptr, EV, F_ACC);
  k_gather<<<nb(ED), 256, 0, stream>>>(t0b, netp, jx, EV, DIMV);
  gemm(t0b, 7, nullptr, t1b, EV, F_RELU);
  gemm(t1b, 8, netp, netb, EV, F_ACC);          // also snapshot net as bf16

  // ---- SoftAgg over kk (= e mod 96) ----
  gemm(netb, 9, t0, nullptr, EV, 0);            // fx fp32
  gemm(netb, 10, t3, nullptr, EV, 0);           // gx fp32
  k_segmax<<<nb(MV * DIMV), 256, 0, stream>>>(t3, smax);
  k_segsum<<<nb(MV * DIMV), 256, 0, stream>>>(t3, t0, smax, yb);
  gemm(yb, 11, hy, nullptr, MV, 0);             // A padded to 128 rows
  k_bcast_add<<<nb(ED), 256, 0, stream>>>(netp, netb, hy, ED);

  // ---- transformer encoder layer over groups of 96 ----
  gemm(netb, 12, t2, t2b, EV, 0);               // x fp32 + bf16
  gemm(t2b, 13, nullptr, t1b, EV, 0);           // qkv bf16-only
  k_attn<<<GV * NHV, 256, 0, stream>>>(t1b, ob);
  gemm(ob, 14, t1, nullptr, EV, 0);             // out_proj(o) fp32
  ln(t2, t2b, t2, t1, PP(45), PP(46), EV, TDIMV, 0);   // x = ln1(x + proj)
  gemm(t2b, 15, nullptr, t0b, EV, F_RELU);      // relu(lin1(x)) bf16-only [E,2048]
  gemm(t0b, 16, t1, nullptr, EV, 0);            // lin2 fp32
  ln(nullptr, t2b, t2, t1, PP(51), PP(52), EV, TDIMV, 0); // x = ln2(x + ff)
  gemm(t2b, 17, netp, nullptr, EV, F_ACC);      // net += t_out(x)

  // ---- GRU stack ----
  for (int l = 0; l < 3; ++l) {
    int pb = 55 + l * 8;
    ln(t0, t0b, netp, nullptr, PP(pb), PP(pb + 1), EV, DIMV, 0);  // h fp32 + bf16
    gemm(t0b, 18 + l * 3, t1, nullptr, EV, F_SIG);                // gate
    gemm(t0b, 19 + l * 3, nullptr, t2b, EV, F_RELU);              // hidden bf16-only
    gemm(t2b, 20 + l * 3, t1 + (size_t)EV * DIMV, nullptr, EV, 0);// res
    k_gru_comb<<<nb(ED), 256, 0, stream>>>(netp, t0, t1,
                                           t1 + (size_t)EV * DIMV, ED);
  }

  // ---- output heads ----
  k_heads<<<(EV + 7) / 8, 256, 0, stream>>>(netp, PP(79), PP(80), PP(81), PP(82),
                                            PP(83), PP(84), netp);
#undef PP
}
